// MultiHeadSelfAttention_5875515261041
// MI455X (gfx1250) — matrix operand order repaired and checked by reading, not compiled, not run on hardware
//
#include <hip/hip_runtime.h>

#define BS  4
#define SEQ 2048
#define DIM 1024
#define NH  16
#define HD  64

typedef __attribute__((ext_vector_type(16))) __bf16 v16bf;
typedef __attribute__((ext_vector_type(8)))  float  v8f;
typedef int v4i __attribute__((vector_size(16)));
typedef unsigned int u4i __attribute__((vector_size(16)));
typedef int v8i __attribute__((vector_size(32)));

union AF { v16bf v; uint4 q[2]; };

static __device__ inline unsigned pack_bf16(float a, float b) {
    unsigned ua = (__float_as_uint(a) + 0x8000u) >> 16;
    unsigned ub = (__float_as_uint(b) + 0x8000u) >> 16;
    return ua | (ub << 16);
}

// ---------------------------------------------------------------------------
// CDNA5 async global->LDS staging (ASYNCcnt-tracked), with sync fallback.
// ---------------------------------------------------------------------------
#if __has_builtin(__builtin_amdgcn_global_load_async_to_lds_b128)
#define HAVE_ASYNC_LDS 1
#endif

static __device__ inline void async_b128(const void* g, void* l) {
#if defined(HAVE_ASYNC_LDS)
    __builtin_amdgcn_global_load_async_to_lds_b128(
        (__attribute__((address_space(1))) v4i*)(v4i*)(void*)g,
        (__attribute__((address_space(3))) v4i*)(v4i*)l, 0, 0);
#else
    *(uint4*)l = *(const uint4*)g;
#endif
}

static __device__ inline void wait_async() {
#if __has_builtin(__builtin_amdgcn_s_wait_asynccnt)
    __builtin_amdgcn_s_wait_asynccnt(0);
#else
    asm volatile("s_wait_asynccnt 0x0" ::: "memory");
#endif
}

// ---------------------------------------------------------------------------
// CDNA5 Tensor Data Mover: 2-D bf16 tile, global -> LDS (TENSORcnt-tracked).
// D# layout per CDNA5 ISA ch.8: group0 = {count/flags, lds_addr, global_addr,
// type=2}; group1 = {data_size, tensor dims, tile dims, dim0 stride}.
// This toolchain exposes the 6-arg builtin (g0, g1, g2, g3, g4, cpol).
// ---------------------------------------------------------------------------
#if __has_builtin(__builtin_amdgcn_tensor_load_to_lds) && \
    __has_builtin(__builtin_amdgcn_s_wait_tensorcnt)
#define HAVE_TDM 1
#endif

static __device__ inline void tdm_load_2d_bf16(const __bf16* gsrc, __bf16* ldst,
                                               unsigned rows, unsigned cols,
                                               unsigned row_stride_elems) {
#if defined(HAVE_TDM)
    const unsigned long long ga = (unsigned long long)gsrc;        // byte addr
    const unsigned la = (unsigned)(unsigned long long)ldst;        // LDS bytes
    u4i g0 = { 1u,                                                 // count=1
               la,                                                 // lds_addr
               (unsigned)ga,                                       // gaddr lo
               (unsigned)(((ga >> 32) & 0x1FFFFFFu) | (2u << 30)) };// hi | type=2
    v8i g1 = { (int)(1u << 16),                                    // data_size=2B
               (int)((cols & 0xFFFFu) << 16),                      // t_dim0 lo
               (int)(((cols >> 16) & 0xFFFFu) | ((rows & 0xFFFFu) << 16)), // t_dim0 hi | t_dim1 lo
               (int)(((rows >> 16) & 0xFFFFu) | ((cols & 0xFFFFu) << 16)), // t_dim1 hi | tile_dim0
               (int)(rows & 0xFFFFu),                              // tile_dim1 (tile_dim2=0)
               (int)row_stride_elems,                              // dim0 stride lo32
               0, 0 };                                             // stride hi, dim1 stride
    v4i gz4 = { 0, 0, 0, 0 };
    v8i gz8 = { 0, 0, 0, 0, 0, 0, 0, 0 };
    __builtin_amdgcn_tensor_load_to_lds(g0, g1, gz4, gz4, gz8, 0);
#else
    (void)gsrc; (void)ldst; (void)rows; (void)cols; (void)row_stride_elems;
#endif
}

static __device__ inline void wait_tensor() {
#if defined(HAVE_TDM)
    __builtin_amdgcn_s_wait_tensorcnt(0);
#endif
}

// ---------------------------------------------------------------------------
// GEMM: out = X (MxK) * W^T (W is NxK row-major) + bias
// TIN = float (convert to bf16 on the fly) or __bf16 (TDM / async staging).
// HEADS_OUT: write bf16 with (b, h, s, hd) layout, times `scale`.
// else:      write f32 row-major (MxN).
// ---------------------------------------------------------------------------
#define BM 128
#define BN 128
#define BKT 32

template <typename TIN, bool HEADS_OUT>
__global__ __launch_bounds__(256) void gemm_wmma(
    const TIN* __restrict__ X, const float* __restrict__ W,
    const float* __restrict__ bias, void* __restrict__ out,
    int M, int N, int K, float scale)
{
    __shared__ __bf16 As[BM][BKT];   // [m][k]
    __shared__ __bf16 Bs[BN][BKT];   // [n][k]  (rows of W)

    const int tid  = threadIdx.x;
    const int lane = tid & 31;
    const int wave = tid >> 5;      // 0..7
    const int wm   = wave & 3;      // 4 sub-tiles of 32 rows
    const int wn   = wave >> 2;     // 2 sub-tiles of 64 cols
    const int hf   = lane >> 4;     // lane half
    const int l16  = lane & 15;

    const int m0 = blockIdx.y * BM;
    const int n0 = blockIdx.x * BN;

    v8f acc[2][4];
#pragma unroll
    for (int i = 0; i < 2; ++i)
#pragma unroll
        for (int j = 0; j < 4; ++j)
            acc[i][j] = (v8f){0.f,0.f,0.f,0.f,0.f,0.f,0.f,0.f};

    // loader: 256 threads, each owns 16 contiguous k of one row (2 threads/row)
    const int lrow = tid >> 1;
    const int lk   = (tid & 1) * 16;

    for (int k0 = 0; k0 < K; k0 += BKT) {
        // ---- stage A tile ----
        {
            const TIN* src = X + (size_t)(m0 + lrow) * K + k0 + lk;
            if constexpr (sizeof(TIN) == 4) {
                const float4* s4 = (const float4*)src;
                float4 f0 = s4[0], f1 = s4[1], f2 = s4[2], f3 = s4[3];
                uint4 p0, p1;
                p0.x = pack_bf16(f0.x, f0.y); p0.y = pack_bf16(f0.z, f0.w);
                p0.z = pack_bf16(f1.x, f1.y); p0.w = pack_bf16(f1.z, f1.w);
                p1.x = pack_bf16(f2.x, f2.y); p1.y = pack_bf16(f2.z, f2.w);
                p1.z = pack_bf16(f3.x, f3.y); p1.w = pack_bf16(f3.z, f3.w);
                uint4* dst = (uint4*)&As[lrow][lk];
                dst[0] = p0; dst[1] = p1;
            } else {
#if defined(HAVE_TDM)
                // one TDM descriptor (wave 0) pulls the whole 128x32 tile
                if (tid < 32)
                    tdm_load_2d_bf16((const __bf16*)X + (size_t)m0 * K + k0,
                                     &As[0][0], BM, BKT, (unsigned)K);
#else
                async_b128(src,     &As[lrow][lk]);
                async_b128(src + 8, &As[lrow][lk] + 8);
#endif
            }
        }
        // ---- stage B tile (always f32 weights) ----
        {
            const float4* s4 = (const float4*)(W + (size_t)(n0 + lrow) * K + k0 + lk);
            float4 f0 = s4[0], f1 = s4[1], f2 = s4[2], f3 = s4[3];
            uint4 p0, p1;
            p0.x = pack_bf16(f0.x, f0.y); p0.y = pack_bf16(f0.z, f0.w);
            p0.z = pack_bf16(f1.x, f1.y); p0.w = pack_bf16(f1.z, f1.w);
            p1.x = pack_bf16(f2.x, f2.y); p1.y = pack_bf16(f2.z, f2.w);
            p1.z = pack_bf16(f3.x, f3.y); p1.w = pack_bf16(f3.z, f3.w);
            uint4* dst = (uint4*)&Bs[lrow][lk];
            dst[0] = p0; dst[1] = p1;
        }
        if (k0 + BKT < K) {
            __builtin_prefetch(X + (size_t)(m0 + lrow) * K + k0 + BKT + lk, 0, 1);
            __builtin_prefetch(W + (size_t)(n0 + lrow) * K + k0 + BKT + lk, 0, 1);
        }
        if constexpr (sizeof(TIN) == 2) {
#if defined(HAVE_TDM)
            wait_tensor();
#else
            wait_async();
#endif
        }
        __syncthreads();

        // ---- compute: 2 m-tiles x 4 n-tiles of 16x16, K=32 ----
        AF a[2];
#pragma unroll
        for (int mt = 0; mt < 2; ++mt) {
            const __bf16* ap = &As[wm * 32 + mt * 16 + l16][hf * 8];
            a[mt].q[0] = *(const uint4*)ap;
            a[mt].q[1] = *(const uint4*)(ap + 16);
        }
#pragma unroll
        for (int nt = 0; nt < 4; ++nt) {
            AF bf;
            const __bf16* bp = &Bs[wn * 64 + nt * 16 + l16][hf * 8];
            bf.q[0] = *(const uint4*)bp;
            bf.q[1] = *(const uint4*)(bp + 16);
#pragma unroll
            for (int mt = 0; mt < 2; ++mt)
                acc[mt][nt] = __builtin_amdgcn_wmma_f32_16x16x32_bf16(
                    false, a[mt].v, false, bf.v, (short)0, acc[mt][nt], false, false);
        }
        __syncthreads();
    }

    // ---- epilogue ----
#pragma unroll
    for (int mt = 0; mt < 2; ++mt) {
#pragma unroll
        for (int nt = 0; nt < 4; ++nt) {
            const int n = n0 + wn * 64 + nt * 16 + l16;
            const float bias_n = bias[n];
#pragma unroll
            for (int r = 0; r < 8; ++r) {
                const int m = m0 + wm * 32 + mt * 16 + r + hf * 8;
                const float val = (acc[mt][nt][r] + bias_n) * scale;
                if constexpr (HEADS_OUT) {
                    const int b = m / SEQ, s = m % SEQ;
                    const int h = n / HD,  d = n % HD;
                    ((__bf16*)out)[(((size_t)(b * NH + h)) * SEQ + s) * HD + d] = (__bf16)val;
                } else {
                    ((float*)out)[(size_t)m * N + n] = val;
                }
            }
        }
    }
}

// ---------------------------------------------------------------------------
// Flash-attention: one block = 128 q-rows of one (b, h). 8 waves x 16 rows.
// Double-buffered staging: K via TDM (TENSORcnt), mask via async-to-LDS
// (ASYNCcnt), V via register prefetch + transposed ds stores (LOAD/DScnt).
// Online softmax, bf16 WMMA throughout.
// ---------------------------------------------------------------------------
#define BQ 128
#define BKB 64

__global__ __launch_bounds__(256) void attn_wmma(
    const __bf16* __restrict__ Q, const __bf16* __restrict__ K,
    const __bf16* __restrict__ V, const int* __restrict__ mask,
    __bf16* __restrict__ ctx)
{
    __shared__ __bf16 Ks[2][BKB][HD];     // [buf][key][d]
    __shared__ __bf16 Vt[2][HD][BKB];     // [buf][d][key]  (transposed)
    __shared__ __bf16 Ps[8][16][BKB];     // per-wave P scratch
    __shared__ int    Mi[2][BKB];         // [buf] raw mask ints (async DMA)

    const int tid  = threadIdx.x;
    const int lane = tid & 31;
    const int wave = tid >> 5;
    const int hf   = lane >> 4;
    const int l16  = lane & 15;

    const int bh = blockIdx.y;
    const int b  = bh / NH;
    const int q0 = blockIdx.x * BQ;

    const __bf16* Qh = Q + (size_t)bh * SEQ * HD;
    const __bf16* Kh = K + (size_t)bh * SEQ * HD;
    const __bf16* Vh = V + (size_t)bh * SEQ * HD;
    const int*    Mh = mask + b * SEQ;

    // Q A-fragments for this wave's 16 rows (d = 0..63 -> 2 frags of K=32)
    AF qf[2];
    {
        const __bf16* qrow = Qh + (size_t)(q0 + wave * 16 + l16) * HD;
#pragma unroll
        for (int kc = 0; kc < 2; ++kc) {
            qf[kc].q[0] = *(const uint4*)(qrow + kc * 32 + hf * 8);
            qf[kc].q[1] = *(const uint4*)(qrow + kc * 32 + hf * 8 + 16);
        }
    }

    v8f o[4];
#pragma unroll
    for (int i = 0; i < 4; ++i) o[i] = (v8f){0.f,0.f,0.f,0.f,0.f,0.f,0.f,0.f};
    float rm[8], rl[8];
#pragma unroll
    for (int r = 0; r < 8; ++r) { rm[r] = -3.0e38f; rl[r] = 0.f; }

    // staging role: 4 threads per key-row, 16 d each
    const int row  = tid >> 2;          // 0..63
    const int dseg = (tid & 3) * 16;    // 0/16/32/48

    union VV { uint4 q[2]; __bf16 e[16]; };

    // ---- prologue: stage block 0 into buffer 0 ----
    {
#if defined(HAVE_TDM)
        if (tid < 32)
            tdm_load_2d_bf16(Kh, &Ks[0][0][0], BKB, HD, HD);
#else
        const __bf16* ksrc = Kh + (size_t)row * HD + dseg;
        async_b128(ksrc,     &Ks[0][row][dseg]);
        async_b128(ksrc + 8, &Ks[0][row][dseg] + 8);
#endif
        // mask block: raw 256B copy via async DMA (16 lanes x B128)
        if (tid < 16)
            async_b128(Mh + tid * 4, &Mi[0][tid * 4]);

        VV vv;
        const uint4* vsrc = (const uint4*)(Vh + (size_t)row * HD + dseg);
        vv.q[0] = vsrc[0]; vv.q[1] = vsrc[1];
#pragma unroll
        for (int i = 0; i < 16; ++i) Vt[0][dseg + i][row] = vv.e[i];
    }
    wait_tensor();
    wait_async();
    __syncthreads();

    int p = 0;
    for (int kb = 0; kb < SEQ; kb += BKB, p ^= 1) {
        const bool more = (kb + BKB) < SEQ;

        // ---- issue next block: K via TDM, mask via async, V into regs ----
        VV vv = {};
        if (more) {
#if defined(HAVE_TDM)
            if (tid < 32)
                tdm_load_2d_bf16(Kh + (size_t)(kb + BKB) * HD,
                                 &Ks[p ^ 1][0][0], BKB, HD, HD);
#else
            const __bf16* ksrc = Kh + (size_t)(kb + BKB + row) * HD + dseg;
            async_b128(ksrc,     &Ks[p ^ 1][row][dseg]);
            async_b128(ksrc + 8, &Ks[p ^ 1][row][dseg] + 8);
#endif
            if (tid < 16)
                async_b128(Mh + kb + BKB + tid * 4, &Mi[p ^ 1][tid * 4]);
            const uint4* vsrc = (const uint4*)(Vh + (size_t)(kb + BKB + row) * HD + dseg);
            vv.q[0] = vsrc[0]; vv.q[1] = vsrc[1];
        }

        // ---- S = Q K^T (4 tiles of 16 keys) ----
        v8f s[4];
#pragma unroll
        for (int nt = 0; nt < 4; ++nt) {
            v8f z = (v8f){0.f,0.f,0.f,0.f,0.f,0.f,0.f,0.f};
#pragma unroll
            for (int kc = 0; kc < 2; ++kc) {
                AF kf;
                const __bf16* kp = &Ks[p][nt * 16 + l16][kc * 32 + hf * 8];
                kf.q[0] = *(const uint4*)kp;
                kf.q[1] = *(const uint4*)(kp + 16);
                z = __builtin_amdgcn_wmma_f32_16x16x32_bf16(
                    false, qf[kc].v, false, kf.v, (short)0, z, false, false);
            }
            const float mb = (Mi[p][nt * 16 + l16] == 0) ? -1.0e30f : 0.0f;
#pragma unroll
            for (int r = 0; r < 8; ++r) z[r] += mb;
            s[nt] = z;
        }

        // ---- online softmax (rows live in register index r + lane half) ----
        float nm[8], sf[8];
#pragma unroll
        for (int r = 0; r < 8; ++r) {
            float mx = fmaxf(fmaxf(s[0][r], s[1][r]), fmaxf(s[2][r], s[3][r]));
#pragma unroll
            for (int off = 1; off < 16; off <<= 1)
                mx = fmaxf(mx, __shfl_xor(mx, off, 32));
            nm[r] = fmaxf(rm[r], mx);
            sf[r] = __expf(rm[r] - nm[r]);
        }
#pragma unroll
        for (int nt = 0; nt < 4; ++nt) {
#pragma unroll
            for (int r = 0; r < 8; ++r) {
                const float pv = __expf(s[nt][r] - nm[r]);
                s[nt][r] = pv;
                Ps[wave][r + hf * 8][nt * 16 + l16] = (__bf16)pv;
            }
        }
#pragma unroll
        for (int r = 0; r < 8; ++r) {
            float rs = s[0][r] + s[1][r] + s[2][r] + s[3][r];
#pragma unroll
            for (int off = 1; off < 16; off <<= 1)
                rs += __shfl_xor(rs, off, 32);
            rl[r] = rl[r] * sf[r] + rs;
            rm[r] = nm[r];
#pragma unroll
            for (int i = 0; i < 4; ++i) o[i][r] *= sf[r];
        }
        // make this wave's P stores visible to its own fragment reloads
        asm volatile("s_wait_dscnt 0x0" ::: "memory");

        // ---- O += P V ----
#pragma unroll
        for (int kc = 0; kc < 2; ++kc) {
            AF pf;
            const __bf16* pp = &Ps[wave][l16][kc * 32 + hf * 8];
            pf.q[0] = *(const uint4*)pp;
            pf.q[1] = *(const uint4*)(pp + 16);
#pragma unroll
            for (int dt = 0; dt < 4; ++dt) {
                AF vf;
                const __bf16* vp = &Vt[p][dt * 16 + l16][kc * 32 + hf * 8];
                vf.q[0] = *(const uint4*)vp;
                vf.q[1] = *(const uint4*)(vp + 16);
                o[dt] = __builtin_amdgcn_wmma_f32_16x16x32_bf16(
                    false, pf.v, false, vf.v, (short)0, o[dt], false, false);
            }
        }

        // ---- commit next block's V into the other buffer ----
        if (more) {
#pragma unroll
            for (int i = 0; i < 16; ++i) Vt[p ^ 1][dseg + i][row] = vv.e[i];
        }
        wait_tensor();
        wait_async();
        __syncthreads();
    }

    // ---- normalize and write context (b, s, h*HD+hd) as bf16 ----
    const int h = bh % NH;
#pragma unroll
    for (int dt = 0; dt < 4; ++dt) {
#pragma unroll
        for (int r = 0; r < 8; ++r) {
            const float val = o[dt][r] / fmaxf(rl[r], 1e-20f);
            const int srow = q0 + wave * 16 + r + hf * 8;
            const int d    = dt * 16 + l16;
            ctx[((size_t)(b * SEQ + srow)) * DIM + h * HD + d] = (__bf16)val;
        }
    }
}

// ---------------------------------------------------------------------------
extern "C" void kernel_launch(void* const* d_in, const int* in_sizes, int n_in,
                              void* d_out, int out_size, void* d_ws, size_t ws_size,
                              hipStream_t stream) {
    const float* q_in = (const float*)d_in[0];
    const float* k_in = (const float*)d_in[1];
    const float* v_in = (const float*)d_in[2];
    const int*   msk  = (const int*)d_in[3];
    const float* Wq = (const float*)d_in[4];
    const float* bq = (const float*)d_in[5];
    const float* Wk = (const float*)d_in[6];
    const float* bk = (const float*)d_in[7];
    const float* Wv = (const float*)d_in[8];
    const float* bv = (const float*)d_in[9];
    const float* Wo = (const float*)d_in[10];
    const float* bo = (const float*)d_in[11];

    const size_t per = (size_t)BS * NH * SEQ * HD;  // 8M elems
    __bf16* Qp  = (__bf16*)d_ws;
    __bf16* Kp  = Qp + per;
    __bf16* Vp  = Kp + per;
    __bf16* Ctx = Vp + per;

    const int M = BS * SEQ;                         // 8192
    dim3 gg(DIM / BN, M / BM);                      // (8, 64)

    gemm_wmma<float, true><<<gg, 256, 0, stream>>>(q_in, Wq, bq, Qp, M, DIM, DIM, 0.125f);
    gemm_wmma<float, true><<<gg, 256, 0, stream>>>(k_in, Wk, bk, Kp, M, DIM, DIM, 1.0f);
    gemm_wmma<float, true><<<gg, 256, 0, stream>>>(v_in, Wv, bv, Vp, M, DIM, DIM, 1.0f);

    attn_wmma<<<dim3(SEQ / BQ, BS * NH), 256, 0, stream>>>(Qp, Kp, Vp, msk, Ctx);

    gemm_wmma<__bf16, false><<<gg, 256, 0, stream>>>(Ctx, Wo, bo, (float*)d_out, M, DIM, DIM, 1.0f);
}
